// Attention_8108898255425
// MI455X (gfx1250) — matrix operand order repaired and checked by reading, not compiled, not run on hardware
//
#include <hip/hip_runtime.h>
#include <hip/hip_bf16.h>
#include <stdint.h>

#define HH 4
#define NN 4096
#define DD 128
#define NEGV (-1.0e30f)

typedef __bf16 bf16;
typedef __attribute__((ext_vector_type(16))) __bf16 v16bf;
typedef __attribute__((ext_vector_type(8)))  float  v8f;

union FragBF { v16bf v; uint4 q[2]; };

__device__ __forceinline__ uint4 ld128(const bf16* p) {
  return *(const uint4*)p;
}
__device__ __forceinline__ bf16 tobf(float f) { return (bf16)f; }

// Pre-pass: convert x,W to bf16 and also produce x^T (d-major) in bf16 so all
// WMMA fragment loads in the main kernel are contiguous b128s.
__global__ void prep_kernel(const float* __restrict__ x, const float* __restrict__ W,
                            bf16* __restrict__ xbf, bf16* __restrict__ wbf,
                            bf16* __restrict__ xT) {
  int i = blockIdx.x * blockDim.x + threadIdx.x;   // < H*N*D
  int d = i % DD;
  int n = (i / DD) % NN;
  int h = i / (DD * NN);
  float xv = x[i];
  xbf[i] = tobf(xv);
  wbf[i] = tobf(W[i]);
  xT[((size_t)h * DD + d) * NN + n] = tobf(xv);
}

// Fused flash-style GAT attention.
// Block = 128 threads = 4 waves; wave w handles head w, rows n0..n0+15.
__global__ __launch_bounds__(128) void gat_kernel(
    const bf16* __restrict__ xbf, const bf16* __restrict__ wbf,
    const bf16* __restrict__ xT, const float* __restrict__ bias,
    const uint8_t* __restrict__ msk, float* __restrict__ out)
{
  __shared__ uint32_t maskw[16][129];   // padded: 16 rows x 128 words of mask bits

  const int tid   = threadIdx.x;
  const int n0    = blockIdx.x * 16;
  const int h     = tid >> 5;
  const int lane  = tid & 31;
  const int col16 = lane & 15;
  const int hlf   = lane >> 4;

  // ---- pack neighbor_mask rows n0..n0+15 into 32-bit words (shared by 4 heads)
  #pragma unroll
  for (int r = 0; r < 16; ++r) {
    const uint4* mp = (const uint4*)(msk + (size_t)(n0 + r) * NN + tid * 32);
    uint4 va = mp[0];
    uint4 vb = mp[1];
    uint32_t w8[8] = {va.x, va.y, va.z, va.w, vb.x, vb.y, vb.z, vb.w};
    uint32_t bits = 0;
    #pragma unroll
    for (int q = 0; q < 8; ++q) {
      uint32_t v = w8[q];
      bits |= (((v & 1u) | ((v >> 7) & 2u) | ((v >> 14) & 4u) | ((v >> 21) & 8u)) << (q * 4));
    }
    maskw[r][tid] = bits;
  }
  __syncthreads();

  // ---- resident B fragments of X^T for the score GEMM (K = d, 4 chunks of 32)
  FragBF bx[4];
  {
    const bf16* xp = xbf + ((size_t)h * NN + n0 + col16) * DD;
    #pragma unroll
    for (int kc = 0; kc < 4; ++kc) {
      bx[kc].q[0] = ld128(xp + 32 * kc + 16 * hlf);
      bx[kc].q[1] = ld128(xp + 32 * kc + 16 * hlf + 8);
    }
  }

  v8f oacc[8];
  #pragma unroll
  for (int dt = 0; dt < 8; ++dt) {
    v8f z = {0.f, 0.f, 0.f, 0.f, 0.f, 0.f, 0.f, 0.f};
    oacc[dt] = z;
  }
  float runM = NEGV, runL = 0.0f;

  const float* bh    = bias + h * NN;
  const bf16* wbase  = wbf + (size_t)h * NN * DD;
  const bf16* xtbase = xT  + (size_t)h * DD * NN;

  for (int m0 = 0; m0 < NN; m0 += 32) {
    // ---- S^T tiles (rows m = m0..m0+31, cols n = n0..n0+15), C initialized with bias
    v8f s[2];
    #pragma unroll
    for (int t = 0; t < 2; ++t) {
      const float4 ba = *(const float4*)(bh + m0 + 16 * t + 8 * hlf);
      const float4 bb = *(const float4*)(bh + m0 + 16 * t + 8 * hlf + 4);
      v8f acc = {ba.x, ba.y, ba.z, ba.w, bb.x, bb.y, bb.z, bb.w};
      const bf16* wp = wbase + (size_t)(m0 + 16 * t + col16) * DD;
      #pragma unroll
      for (int kc = 0; kc < 4; ++kc) {
        FragBF aw;
        aw.q[0] = ld128(wp + 32 * kc + 16 * hlf);
        aw.q[1] = ld128(wp + 32 * kc + 16 * hlf + 8);
        acc = __builtin_amdgcn_wmma_f32_16x16x32_bf16(
            false, aw.v, false, bx[kc].v, (short)0, acc, false, false);
      }
      s[t] = acc;
    }

    // ---- leaky-relu, mask, online softmax (row = this lane's col16)
    const uint32_t mw = maskw[col16][m0 >> 5];
    float sc[16];
    float cmax = NEGV;
    #pragma unroll
    for (int t = 0; t < 2; ++t) {
      #pragma unroll
      for (int j = 0; j < 8; ++j) {
        float v = s[t][j];
        v = v > 0.0f ? v : 0.2f * v;                       // LeakyReLU(0.2)
        const bool on = (mw >> (16 * t + 8 * hlf + j)) & 1;
        v = on ? v : NEGV;
        sc[t * 8 + j] = v;
        cmax = fmaxf(cmax, v);
      }
    }
    cmax = fmaxf(cmax, __shfl_xor(cmax, 16, 32));          // combine halves of the row
    const float mnew  = fmaxf(runM, cmax);
    const float alpha = __expf(runM - mnew);
    runM = mnew;

    float psum = 0.0f;
    FragBF pf;                                             // P as 16x32 bf16 A-fragment
    #pragma unroll
    for (int t = 0; t < 2; ++t) {
      #pragma unroll
      for (int j = 0; j < 8; ++j) {
        const bool on = (mw >> (16 * t + 8 * hlf + j)) & 1;
        float e = on ? __expf(sc[t * 8 + j] - mnew) : 0.0f;
        psum += e;
        pf.v[t * 8 + j] = tobf(e);
      }
    }
    psum += __shfl_xor(psum, 16, 32);
    runL = runL * alpha + psum;

    // ---- rescale O accumulator: broadcast alpha of row (j + 8*hlf)
    float arow[8];
    #pragma unroll
    for (int j = 0; j < 8; ++j) arow[j] = __shfl(alpha, j + 8 * hlf, 32);
    #pragma unroll
    for (int dt = 0; dt < 8; ++dt) {
      #pragma unroll
      for (int j = 0; j < 8; ++j) oacc[dt][j] *= arow[j];
    }

    // ---- O += P(16x32) x Xm(32x128); Xm B-fragments from contiguous xT rows
    #pragma unroll
    for (int dt = 0; dt < 8; ++dt) {
      FragBF bxm;
      const bf16* tp = xtbase + (size_t)(dt * 16 + col16) * NN + m0 + 8 * hlf;
      bxm.q[0] = ld128(tp);
      bxm.q[1] = ld128(tp + 16);
      oacc[dt] = __builtin_amdgcn_wmma_f32_16x16x32_bf16(
          false, pf.v, false, bxm.v, (short)0, oacc[dt], false, false);
    }

    if (m0 + 32 < NN) {
      __builtin_prefetch(wbase + (size_t)(m0 + 32 + col16) * DD, 0, 1);
    }
  }

  // ---- normalize by row sums and store
  const float linv = __builtin_amdgcn_rcpf(runL);
  float rrow[8];
  #pragma unroll
  for (int j = 0; j < 8; ++j) rrow[j] = __shfl(linv, j + 8 * hlf, 32);

  float* op = out + ((size_t)h * NN + n0 + 8 * hlf) * DD + col16;
  #pragma unroll
  for (int j = 0; j < 8; ++j) {
    #pragma unroll
    for (int dt = 0; dt < 8; ++dt) {
      op[(size_t)j * DD + dt * 16] = oacc[dt][j] * rrow[j];
    }
  }
}

extern "C" void kernel_launch(void* const* d_in, const int* in_sizes, int n_in,
                              void* d_out, int out_size, void* d_ws, size_t ws_size,
                              hipStream_t stream) {
  (void)in_sizes; (void)n_in; (void)out_size; (void)ws_size;
  const float*   x   = (const float*)d_in[0];
  const float*   W   = (const float*)d_in[1];
  const float*   b   = (const float*)d_in[2];
  const uint8_t* msk = (const uint8_t*)d_in[3];
  float*         out = (float*)d_out;

  bf16* xbf = (bf16*)d_ws;                       //  4 MB
  bf16* wbf = xbf + (size_t)HH * NN * DD;        //  4 MB
  bf16* xT  = wbf + (size_t)HH * NN * DD;        //  4 MB

  const int total = HH * NN * DD;                // 2,097,152
  prep_kernel<<<total / 256, 256, 0, stream>>>(x, W, xbf, wbf, xT);
  gat_kernel<<<NN / 16, 128, 0, stream>>>(xbf, wbf, xT, b, msk, out);
}
